// ScaledDotProductAttention_13262859010784
// MI455X (gfx1250) — matrix operand order repaired and checked by reading, not compiled, not run on hardware
//
#include <hip/hip_runtime.h>

// MI455X (gfx1250) fused attention: S = QK^T/(sqrt(256)*20), attn = softmax(S),
// ctx = attn @ V.  Outputs BOTH ctx and attn (d_out = [ctx | attn], fp32).
// Two-pass flash style: pass1 row stats, pass2 recompute + stream attn + P@V.
// f16 WMMA (v_wmma_f32_16x16x32_f16) with fp32 accumulation.
// v2: double-buffered LDS staging (1 barrier/iter) + global_prefetch_b8 of the
//     next tile overlapped with the WMMA phase.

typedef __attribute__((ext_vector_type(16))) _Float16 v16h;
typedef __attribute__((ext_vector_type(8)))  _Float16 v8h;
typedef __attribute__((ext_vector_type(8)))  float    v8f;

#define B_    16
#define L_    2048
#define D_    256
#define QTILE 128          // query rows per workgroup (8 waves x 16 rows)
#define NW    8
#define KSTEP 32           // keys per staged LDS tile (= WMMA K for P@V)
#define KPAD  8            // LDS padding (halves); keeps 16B row alignment
#define SCALE (1.0f / 320.0f)   // 1/(sqrt(256) * 20)

// read halves p[0..7] and p[16..23] from LDS as two b128 loads
__device__ __forceinline__ v16h lds_read16(const _Float16* p) {
  v8h lo = *(const v8h*)(p);
  v8h hh = *(const v8h*)(p + 16);
  v16h r;
#pragma unroll
  for (int j = 0; j < 8; ++j) { r[j] = lo[j]; r[j + 8] = hh[j]; }
  return r;
}

// prefetch one 128B staging segment of the next tile (speculative, no VGPR cost)
__device__ __forceinline__ void prefetch_tile(const float* __restrict__ g, int tid) {
  const int row = tid >> 3;
  const int cb  = (tid & 7) * 32;
  const float* p = g + (size_t)row * D_ + cb;
  __builtin_prefetch(p, 0, 3);
  __builtin_prefetch(p + 16, 0, 3);   // cover the full 128B segment
}

// cooperative: 32 rows x 256 cols fp32 -> f16 LDS tile (row-major)
__device__ __forceinline__ void stage_k(const float* __restrict__ gsrc,
                                        _Float16 (*Ks)[D_ + KPAD], int tid) {
  const int row = tid >> 3;           // 0..31
  const int cb  = (tid & 7) * 32;     // 0..224
  const float4* s4 = (const float4*)(gsrc + (size_t)row * D_ + cb);
  _Float16* dst = &Ks[row][cb];
#pragma unroll
  for (int i = 0; i < 4; ++i) {
    float4 a = s4[2 * i], b = s4[2 * i + 1];
    v8h h;
    h[0] = (_Float16)a.x; h[1] = (_Float16)a.y; h[2] = (_Float16)a.z; h[3] = (_Float16)a.w;
    h[4] = (_Float16)b.x; h[5] = (_Float16)b.y; h[6] = (_Float16)b.z; h[7] = (_Float16)b.w;
    *(v8h*)(dst + 8 * i) = h;
  }
}

// cooperative: 32 rows x 256 cols fp32 -> TRANSPOSED f16 LDS tile Vt[d][k]
__device__ __forceinline__ void stage_vt(const float* __restrict__ gsrc,
                                         _Float16 (*Vt)[KSTEP + KPAD], int tid) {
  const int krow = tid >> 3;
  const int cb   = (tid & 7) * 32;
  const float* s = gsrc + (size_t)krow * D_ + cb;
#pragma unroll
  for (int i = 0; i < 32; ++i) Vt[cb + i][krow] = (_Float16)s[i];
}

__global__ void __launch_bounds__(256, 1)
attn_fused(const float* __restrict__ Qg, const float* __restrict__ Kg,
           const float* __restrict__ Vg, float* __restrict__ ctx,
           float* __restrict__ attn) {
  __shared__ __align__(16) _Float16 Ks[2][KSTEP][D_ + KPAD];   // ~33.8 KB (dbl buf)
  __shared__ __align__(16) _Float16 Vt[2][D_][KSTEP + KPAD];   // 40 KB  (dbl buf)
  __shared__ __align__(16) _Float16 Ps[NW][16][KSTEP];         // 8 KB (P transpose)

  const int tid   = threadIdx.x;
  const int lane  = tid & 31;
  const int wv    = tid >> 5;
  const int n     = lane & 15;         // column / key-in-tile index
  const int hi    = lane >> 4;         // half-wave select
  const int b     = blockIdx.y;
  const int qbase = blockIdx.x * QTILE + wv * 16;

  const float* Kb = Kg + (size_t)b * L_ * D_;
  const float* Vb = Vg + (size_t)b * L_ * D_;

  // ---- Q A-fragments (16 rows x 256 d), resident all kernel: 64 VGPRs ----
  // A layout (16-bit, 16x32): lane<16: K = d0+0..7 & d0+16..23 ; lane>=16: +8
  v16h aQ[8];
  {
    const float* qrow = Qg + ((size_t)(b * L_ + qbase + n)) * D_;
#pragma unroll
    for (int c = 0; c < 8; ++c) {
      const int d0 = c * 32;
      const float* p0 = qrow + d0 + hi * 8;
      const float* p1 = qrow + d0 + 16 + hi * 8;
      v16h a;
#pragma unroll
      for (int j = 0; j < 8; ++j) { a[j] = (_Float16)p0[j]; a[j + 8] = (_Float16)p1[j]; }
      aQ[c] = a;
    }
  }

  // ---- pass 1: per-lane flash stats over this lane's columns ----
  // C/D layout: acc VGPR r, lane -> row M = r + 8*hi, col N = n
  float m[8], s[8];
#pragma unroll
  for (int r = 0; r < 8; ++r) { m[r] = -3.0e38f; s[r] = 0.0f; }

  for (int kb = 0; kb < L_; kb += KSTEP) {
    const int p = (kb >> 5) & 1;
    stage_k(Kb + (size_t)kb * D_, Ks[p], tid);
    __syncthreads();
    if (kb + KSTEP < L_)                       // overlap next-tile fetch w/ WMMAs
      prefetch_tile(Kb + (size_t)(kb + KSTEP) * D_, tid);

#pragma unroll
    for (int t = 0; t < 2; ++t) {
      v8f c = {};
#pragma unroll
      for (int ch = 0; ch < 8; ++ch) {
        // B = K^T (32x16): lane n holds key row (kb+16t+n), d = 8hi+0..7 & 16+8hi+0..7
        v16h bk = lds_read16(&Ks[p][t * 16 + n][ch * 32 + hi * 8]);
        c = __builtin_amdgcn_wmma_f32_16x16x32_f16(false, aQ[ch], false, bk,
                                                   (short)0, c, false, false);
      }
#pragma unroll
      for (int r = 0; r < 8; ++r) {
        float x  = c[r] * SCALE;
        float mn = fmaxf(m[r], x);
        s[r] = s[r] * __expf(m[r] - mn) + __expf(x - mn);
        m[r] = mn;
      }
    }
  }

  // warm the caches for pass 2 while we reduce the stats
  prefetch_tile(Kb, tid);
  prefetch_tile(Vb, tid);

  // ---- merge stats across the 16 lanes holding each row (log-sum-exp) ----
  float rz[8];
#pragma unroll
  for (int r = 0; r < 8; ++r) {
    float mm = m[r], ss = s[r];
#pragma unroll
    for (int msk = 1; msk < 16; msk <<= 1) {
      float mo = __shfl_xor(mm, msk, 32);
      float so = __shfl_xor(ss, msk, 32);
      float mn = fmaxf(mm, mo);
      ss = ss * __expf(mm - mn) + so * __expf(mo - mn);
      mm = mn;
    }
    m[r]  = mm;
    rz[r] = 1.0f / ss;
  }

  // make sure every wave is done reading pass-1's last Ks buffer before reuse
  __syncthreads();

  // ---- pass 2: recompute S, stream normalized attn (NT), accumulate P@V ----
  v8f acc[16];
#pragma unroll
  for (int dt = 0; dt < 16; ++dt) acc[dt] = (v8f){};

  for (int kb = 0; kb < L_; kb += KSTEP) {
    const int p = (kb >> 5) & 1;
    stage_k (Kb + (size_t)kb * D_, Ks[p], tid);
    stage_vt(Vb + (size_t)kb * D_, Vt[p], tid);
    __syncthreads();
    if (kb + KSTEP < L_) {                     // overlap next-tile fetch w/ WMMAs
      prefetch_tile(Kb + (size_t)(kb + KSTEP) * D_, tid);
      prefetch_tile(Vb + (size_t)(kb + KSTEP) * D_, tid);
    }

#pragma unroll
    for (int t = 0; t < 2; ++t) {
      v8f c = {};
#pragma unroll
      for (int ch = 0; ch < 8; ++ch) {
        v16h bk = lds_read16(&Ks[p][t * 16 + n][ch * 32 + hi * 8]);
        c = __builtin_amdgcn_wmma_f32_16x16x32_f16(false, aQ[ch], false, bk,
                                                   (short)0, c, false, false);
      }
#pragma unroll
      for (int r = 0; r < 8; ++r) {
        float x = c[r] * SCALE;
        float pv = __expf(x - m[r]) * rz[r];
        const int row = qbase + r + 8 * hi;
        const int col = kb + t * 16 + n;
        __builtin_nontemporal_store(pv, &attn[((size_t)b * L_ + row) * L_ + col]);
        Ps[wv][r + 8 * hi][t * 16 + n] = (_Float16)pv;  // lane-transpose via LDS
      }
    }

    // wave-private LDS RAW (lane exchange); same wave, so a dscnt wait suffices
    asm volatile("s_wait_dscnt 0" ::: "memory");

    // P as A-operand (16x32 f16)
    v16h pA;
    {
      v8h lo = *(const v8h*)&Ps[wv][n][hi * 8];
      v8h hh = *(const v8h*)&Ps[wv][n][16 + hi * 8];
#pragma unroll
      for (int j = 0; j < 8; ++j) { pA[j] = lo[j]; pA[j + 8] = hh[j]; }
    }

#pragma unroll
    for (int dt = 0; dt < 16; ++dt) {
      // B = V (32x16): lane n holds output dim (dt*16+n), keys = 8hi+0..7 & 16+8hi+0..7
      v16h bv = lds_read16(&Vt[p][dt * 16 + n][hi * 8]);
      acc[dt] = __builtin_amdgcn_wmma_f32_16x16x32_f16(false, pA, false, bv,
                                                       (short)0, acc[dt], false, false);
    }
  }

  // ---- write context ----
#pragma unroll
  for (int dt = 0; dt < 16; ++dt) {
#pragma unroll
    for (int r = 0; r < 8; ++r) {
      const int row = qbase + r + 8 * hi;
      ctx[((size_t)b * L_ + row) * D_ + dt * 16 + n] = acc[dt][r];
    }
  }
}

extern "C" void kernel_launch(void* const* d_in, const int* in_sizes, int n_in,
                              void* d_out, int out_size, void* d_ws, size_t ws_size,
                              hipStream_t stream) {
  (void)in_sizes; (void)n_in; (void)out_size; (void)d_ws; (void)ws_size;
  const float* Q = (const float*)d_in[0];
  const float* K = (const float*)d_in[1];
  const float* V = (const float*)d_in[2];
  // d_in[3..6] = num / if_draw / cls / if_att_score scalars; the harness setup
  // fixes if_draw=1, if_att_score=1, which this kernel implements directly.
  float* out  = (float*)d_out;
  float* ctx  = out;                               // B*L*D context first
  float* attn = out + (size_t)B_ * L_ * D_;        // then B*L*L attention

  dim3 grid(L_ / QTILE, B_);
  attn_fused<<<grid, 256, 0, stream>>>(Q, K, V, ctx, attn);
}
